// GCNLayer_25469156065919
// MI455X (gfx1250) — matrix operand order repaired and checked by reading, not compiled, not run on hardware
//
#include <hip/hip_runtime.h>
#include <hip/hip_bf16.h>

#define N_NODES 8192
#define IN_DIM 128
#define THETA_DIM 256
#define OUT_DIM 64

typedef __bf16   v16bf __attribute__((ext_vector_type(16)));
typedef float    v8f   __attribute__((ext_vector_type(8)));
typedef float    v4f   __attribute__((ext_vector_type(4)));
typedef unsigned int v4u __attribute__((ext_vector_type(4)));

union FragB16 {
    v16bf bf;
    v4u   u4[2];
    unsigned int u[8];
};

__device__ __forceinline__ unsigned int f32_to_bf16_bits(float f) {
    unsigned int u = __float_as_uint(f);
    unsigned int r = u + 0x7FFFu + ((u >> 16) & 1u);   // round-to-nearest-even
    return r >> 16;
}
__device__ __forceinline__ unsigned int pack_bf16(float lo, float hi) {
    return f32_to_bf16_bits(lo) | (f32_to_bf16_bits(hi) << 16);
}

// ---------------------------------------------------------------------------
// 1) BatchNorm stats: per-feature mean/var over 8192 rows -> fused scale/shift
// ---------------------------------------------------------------------------
__global__ __launch_bounds__(256) void bn_stats_kernel(
    const float* __restrict__ H, const float* __restrict__ gamma,
    const float* __restrict__ beta, float* __restrict__ sc, float* __restrict__ bc) {
    __shared__ float s_sum[256], s_sq[256];
    int k = blockIdx.x, t = threadIdx.x;
    float s = 0.f, q = 0.f;
    for (int r = t; r < N_NODES; r += 256) {
        float v = H[(size_t)r * IN_DIM + k];
        s += v; q += v * v;
    }
    s_sum[t] = s; s_sq[t] = q;
    __syncthreads();
    for (int w = 128; w > 0; w >>= 1) {
        if (t < w) { s_sum[t] += s_sum[t + w]; s_sq[t] += s_sq[t + w]; }
        __syncthreads();
    }
    if (t == 0) {
        float mean = s_sum[0] * (1.0f / N_NODES);
        float var  = s_sq[0] * (1.0f / N_NODES) - mean * mean;
        float scale = rsqrtf(var + 1e-5f) * gamma[k];
        sc[k] = scale;
        bc[k] = beta[k] - mean * scale;
    }
}

// ---------------------------------------------------------------------------
// 2) Per row: Hn (LDS), Hx = Hn@W1+b1 as bf16, HWout = Hn@Wout+bout as f32
// ---------------------------------------------------------------------------
__global__ __launch_bounds__(256) void hx_kernel(
    const float* __restrict__ H, const float* __restrict__ sc,
    const float* __restrict__ bc, const float* __restrict__ W1,
    const float* __restrict__ b1, const float* __restrict__ Wout,
    const float* __restrict__ bout,
    unsigned short* __restrict__ Hxb, float* __restrict__ HWout) {
    __shared__ float hn[IN_DIM];
    int i = blockIdx.x, t = threadIdx.x;
    if (t < IN_DIM) hn[t] = H[(size_t)i * IN_DIM + t] * sc[t] + bc[t];
    __syncthreads();
    float acc = b1[t];
#pragma unroll 4
    for (int k = 0; k < IN_DIM; ++k) acc = fmaf(hn[k], W1[k * THETA_DIM + t], acc);
    Hxb[(size_t)i * THETA_DIM + t] = (unsigned short)f32_to_bf16_bits(acc);
    if (t < OUT_DIM) {
        float a2 = bout[t];
#pragma unroll 4
        for (int k = 0; k < IN_DIM; ++k) a2 = fmaf(hn[k], Wout[k * OUT_DIM + t], a2);
        HWout[(size_t)i * OUT_DIM + t] = a2;
    }
}

__global__ void zero_kernel(float* __restrict__ p, int n) {
    int i = blockIdx.x * 256 + threadIdx.x;
    if (i < n) p[i] = 0.f;
}

// ---------------------------------------------------------------------------
// 4) Gram GEMM: A1 = max(sigmoid(Hx@Hx^T),0.1)*mask + I ; rowsum accumulation
//    Block: 8 waves -> 128(M) x 64(N) tile; wave: 32x32 = 2x2 WMMA accums.
// ---------------------------------------------------------------------------
__global__ __launch_bounds__(256) void gram_kernel(
    const unsigned short* __restrict__ Hxb,   // [N, 256] bf16
    const float* __restrict__ Amask,          // [N, N]
    float* __restrict__ A1,                   // [N, N] (in d_out)
    float* __restrict__ rowsum) {
    __shared__ float s_row[128];
    int lane = threadIdx.x & 31;
    int w    = threadIdx.x >> 5;
    int wm = w >> 1, wn = w & 1;
    int rowBase = blockIdx.y * 128 + wm * 32;
    int colBase = blockIdx.x * 64  + wn * 32;
    int half = lane >> 4;     // 0|1
    int l15  = lane & 15;

    if (threadIdx.x < 128) s_row[threadIdx.x] = 0.f;
    __syncthreads();

    // Prefetch the HBM-resident mask stream (only true HBM input here).
#pragma unroll
    for (int mt = 0; mt < 2; ++mt)
        __builtin_prefetch(Amask + (size_t)(rowBase + mt * 16 + l15) * N_NODES
                                 + colBase + half * 32, 0, 0);

    v8f acc[2][2] = {};
    const unsigned short* arow[2];
    const unsigned short* brow[2];
#pragma unroll
    for (int mt = 0; mt < 2; ++mt)
        arow[mt] = Hxb + (size_t)(rowBase + mt * 16 + l15) * THETA_DIM;
#pragma unroll
    for (int nt = 0; nt < 2; ++nt)
        brow[nt] = Hxb + (size_t)(colBase + nt * 16 + l15) * THETA_DIM;

    for (int k0 = 0; k0 < THETA_DIM; k0 += 32) {
        FragB16 a[2], b[2];
        int ka = k0 + half * 8;    // A: K split {ka..ka+7, ka+16..ka+23}
        int kb = k0 + half * 8;    // B: K split {kb..kb+7, kb+16..kb+23}
#pragma unroll
        for (int mt = 0; mt < 2; ++mt) {
            a[mt].u4[0] = *(const v4u*)(arow[mt] + ka);
            a[mt].u4[1] = *(const v4u*)(arow[mt] + ka + 16);
        }
#pragma unroll
        for (int nt = 0; nt < 2; ++nt) {
            b[nt].u4[0] = *(const v4u*)(brow[nt] + kb);
            b[nt].u4[1] = *(const v4u*)(brow[nt] + kb + 16);
        }
#pragma unroll
        for (int mt = 0; mt < 2; ++mt)
#pragma unroll
            for (int nt = 0; nt < 2; ++nt)
                acc[mt][nt] = __builtin_amdgcn_wmma_f32_16x16x32_bf16(
                    false, a[mt].bf, false, b[nt].bf, (short)0, acc[mt][nt],
                    false, false);
    }

    // Epilogue: D layout -> m = r + half*8, n = l15
#pragma unroll
    for (int mt = 0; mt < 2; ++mt) {
#pragma unroll
        for (int r = 0; r < 8; ++r) {
            int i = rowBase + mt * 16 + half * 8 + r;
            float part = 0.f;
#pragma unroll
            for (int nt = 0; nt < 2; ++nt) {
                int j = colBase + nt * 16 + l15;
                float score = 1.0f / (1.0f + __expf(-acc[mt][nt][r]));
                float m = ceilf(Amask[(size_t)i * N_NODES + j] * 1e-5f);
                float v = fmaxf(score, 0.1f) * m + ((i == j) ? 1.0f : 0.0f);
                A1[(size_t)i * N_NODES + j] = v;
                part += v;
            }
            atomicAdd(&s_row[wm * 32 + mt * 16 + half * 8 + r], part);  // ds_add_f32
        }
    }
    __syncthreads();
    if (threadIdx.x < 128)
        atomicAdd(&rowsum[blockIdx.y * 128 + threadIdx.x], s_row[threadIdx.x]);
}

// ---------------------------------------------------------------------------
// 5) d = rowsum^-0.5
// ---------------------------------------------------------------------------
__global__ void dvec_kernel(const float* __restrict__ rowsum, float* __restrict__ dvec) {
    int i = blockIdx.x * 256 + threadIdx.x;
    if (i < N_NODES) dvec[i] = rsqrtf(rowsum[i]);
}

// ---------------------------------------------------------------------------
// 6) Gt[c][j] = bf16(d[j] * HWout[j][c])  (folds A_hat column scaling into B)
// ---------------------------------------------------------------------------
__global__ void gt_kernel(const float* __restrict__ HWout,
                          const float* __restrict__ dvec,
                          unsigned short* __restrict__ Gt) {
    int idx = blockIdx.x * 256 + threadIdx.x;   // 64 * 8192 elements
    int c = idx >> 13;
    int j = idx & (N_NODES - 1);
    Gt[(size_t)c * N_NODES + j] =
        (unsigned short)f32_to_bf16_bits(dvec[j] * HWout[(size_t)j * OUT_DIM + c]);
}

// ---------------------------------------------------------------------------
// 7) out = LeakyReLU(d_i * (A1 @ Gt^T))  — bf16 WMMA, A1 converted in-register
//    Block: 8 waves, each wave = 16(M) x 64(N), K loop over 8192.
// ---------------------------------------------------------------------------
__global__ __launch_bounds__(256) void out_kernel(
    const float* __restrict__ A1,            // [N,N] f32
    const unsigned short* __restrict__ Gt,   // [64,N] bf16
    const float* __restrict__ dvec,
    float* __restrict__ out) {
    int lane = threadIdx.x & 31;
    int w    = threadIdx.x >> 5;
    int half = lane >> 4, l15 = lane & 15;
    int rowBase = blockIdx.x * 128 + w * 16;

    v8f acc[4] = {};
    const float* arow = A1 + (size_t)(rowBase + l15) * N_NODES;

    for (int k0 = 0; k0 < N_NODES; k0 += 32) {
        int ka = k0 + half * 8;
        v4f f0 = *(const v4f*)(arow + ka);
        v4f f1 = *(const v4f*)(arow + ka + 4);
        v4f f2 = *(const v4f*)(arow + ka + 16);
        v4f f3 = *(const v4f*)(arow + ka + 20);
        FragB16 a;
        a.u[0] = pack_bf16(f0[0], f0[1]);
        a.u[1] = pack_bf16(f0[2], f0[3]);
        a.u[2] = pack_bf16(f1[0], f1[1]);
        a.u[3] = pack_bf16(f1[2], f1[3]);
        a.u[4] = pack_bf16(f2[0], f2[1]);
        a.u[5] = pack_bf16(f2[2], f2[3]);
        a.u[6] = pack_bf16(f3[0], f3[1]);
        a.u[7] = pack_bf16(f3[2], f3[3]);
        int kb = k0 + half * 8;
#pragma unroll
        for (int nt = 0; nt < 4; ++nt) {
            const unsigned short* brow = Gt + (size_t)(nt * 16 + l15) * N_NODES + kb;
            FragB16 b;
            b.u4[0] = *(const v4u*)(brow);
            b.u4[1] = *(const v4u*)(brow + 16);
            acc[nt] = __builtin_amdgcn_wmma_f32_16x16x32_bf16(
                false, a.bf, false, b.bf, (short)0, acc[nt], false, false);
        }
    }

#pragma unroll
    for (int nt = 0; nt < 4; ++nt) {
#pragma unroll
        for (int r = 0; r < 8; ++r) {
            int i = rowBase + half * 8 + r;
            int c = nt * 16 + l15;
            float v = acc[nt][r] * dvec[i];
            v = (v >= 0.f) ? v : 0.01f * v;
            out[(size_t)i * OUT_DIM + c] = v;
        }
    }
}

// ---------------------------------------------------------------------------
extern "C" void kernel_launch(void* const* d_in, const int* in_sizes, int n_in,
                              void* d_out, int out_size, void* d_ws, size_t ws_size,
                              hipStream_t stream) {
    const float* H     = (const float*)d_in[0];
    const float* A     = (const float*)d_in[1];
    const float* gamma = (const float*)d_in[2];
    const float* beta  = (const float*)d_in[3];
    const float* W1    = (const float*)d_in[4];
    const float* b1    = (const float*)d_in[5];
    const float* Wout  = (const float*)d_in[6];
    const float* bout  = (const float*)d_in[7];
    (void)in_sizes; (void)n_in; (void)out_size; (void)ws_size;

    float* out = (float*)d_out;                                 // [8192,64]
    float* A1  = (float*)d_out + (size_t)N_NODES * OUT_DIM;     // [8192,8192]

    char* ws = (char*)d_ws;
    float* sc            = (float*)(ws + 0);                    // 128 f
    float* bc            = (float*)(ws + 512);                  // 128 f
    float* rowsum        = (float*)(ws + 1024);                 // 8192 f
    float* dvec          = (float*)(ws + 33792);                // 8192 f
    unsigned short* Hxb  = (unsigned short*)(ws + 66560);       // 8192*256 bf16 (4 MB)
    float* HWout         = (float*)(ws + 66560 + 4194304);      // 8192*64 f32 (2 MB)
    unsigned short* Gt   = (unsigned short*)(ws + 66560 + 4194304 + 2097152); // 1 MB

    bn_stats_kernel<<<IN_DIM, 256, 0, stream>>>(H, gamma, beta, sc, bc);
    hx_kernel<<<N_NODES, 256, 0, stream>>>(H, sc, bc, W1, b1, Wout, bout, Hxb, HWout);
    zero_kernel<<<N_NODES / 256, 256, 0, stream>>>(rowsum, N_NODES);

    dim3 g1(N_NODES / 64, N_NODES / 128);
    gram_kernel<<<g1, 256, 0, stream>>>(Hxb, A, A1, rowsum);

    dvec_kernel<<<N_NODES / 256, 256, 0, stream>>>(rowsum, dvec);
    gt_kernel<<<(OUT_DIM * N_NODES) / 256, 256, 0, stream>>>(HWout, dvec, Gt);
    out_kernel<<<N_NODES / 128, 256, 0, stream>>>(A1, Gt, dvec, out);
}
